// Decoder_28819230556826
// MI455X (gfx1250) — compile-verified, matrix operand order checked by reading, not run on hardware
//
#include <hip/hip_runtime.h>
#include <hip/hip_bf16.h>
#include <cstdint>

// ---------------------------------------------------------------------------
// Problem constants (match reference)
// ---------------------------------------------------------------------------
#define BS    4096
#define HID   1024
#define INP   512
#define NSPAN 8
#define NOUT  2
#define TSTEPS 20
#define KX    (INP + HID)   // 1536 : fused K dimension [inp | h]
#define G4    (4 * HID)     // 4096 : gate rows
#define NK    (KX / 32)     // 48 K-chunks of 32
#define LDSTR 40            // LDS row stride in bf16 elems (32 data + 8 pad = 80B)

typedef __bf16 bf16;
typedef bf16  v16bf __attribute__((ext_vector_type(16)));
typedef float v8f   __attribute__((ext_vector_type(8)));

union FragBF {          // 16 bf16 = 32 bytes = two 16B chunks per lane
    v16bf v;
    uint4 q[2];
};

__device__ __forceinline__ float sigm(float x) { return 1.0f / (1.0f + __expf(-x)); }

// CDNA5 async DMA: per-lane global -> LDS, tracked by ASYNCcnt.
__device__ __forceinline__ void async_b128(unsigned lds_addr, const void* gaddr) {
    asm volatile("global_load_async_to_lds_b128 %0, %1, off"
                 :: "v"(lds_addr), "v"(gaddr)
                 : "memory");
}
__device__ __forceinline__ void async_b64(unsigned lds_addr, const void* gaddr) {
    asm volatile("global_load_async_to_lds_b64 %0, %1, off"
                 :: "v"(lds_addr), "v"(gaddr)
                 : "memory");
}
// Wait until my chunk's DMAs landed while the next chunk's 3 are in flight.
__device__ __forceinline__ void wait_async_le3() {
    asm volatile("s_wait_asynccnt 0x3" ::: "memory");
}
__device__ __forceinline__ void wait_async0() {
    asm volatile("s_wait_asynccnt 0x0" ::: "memory");
}

// ---------------------------------------------------------------------------
// Weight prep: fp32 -> bf16, build Wcat = [W_ih | W_hh] (row-major [4H, KX]),
// W_fcin bf16, fused gate bias = b_ih + b_hh.
// ---------------------------------------------------------------------------
__global__ void prep_weights(const float* __restrict__ Wih,
                             const float* __restrict__ Whh,
                             const float* __restrict__ Wfcin_f,
                             const float* __restrict__ bih,
                             const float* __restrict__ bhh,
                             bf16* __restrict__ Wcat,
                             bf16* __restrict__ Wfc,
                             float* __restrict__ biasg) {
    const int idx    = blockIdx.x * blockDim.x + threadIdx.x;
    const int stride = gridDim.x * blockDim.x;
    for (int i = idx; i < G4 * KX; i += stride) {
        const int r = i / KX, k = i % KX;
        const float v = (k < INP) ? Wih[r * INP + k] : Whh[r * HID + (k - INP)];
        Wcat[i] = (bf16)v;
    }
    for (int i = idx; i < INP * HID; i += stride) Wfc[i] = (bf16)Wfcin_f[i];
    for (int i = idx; i < G4; i += stride) biasg[i] = bih[i] + bhh[i];
}

// ---------------------------------------------------------------------------
// State init: h/c fp32 working copies; X0 = [zeros(inp) | bf16(hx)]
// ---------------------------------------------------------------------------
__global__ void init_state(const float* __restrict__ hx,
                           const float* __restrict__ cx,
                           float* __restrict__ h,
                           float* __restrict__ c,
                           bf16* __restrict__ X0) {
    const int idx    = blockIdx.x * blockDim.x + threadIdx.x;
    const int stride = gridDim.x * blockDim.x;
    for (int i = idx; i < BS * HID; i += stride) {
        const float hv = hx[i];
        h[i] = hv;
        c[i] = cx[i];
        const int row = i / HID, col = i % HID;
        X0[(size_t)row * KX + INP + col] = (bf16)hv;
    }
    for (int i = idx; i < BS * INP; i += stride) {
        const int row = i / INP, col = i % INP;
        X0[(size_t)row * KX + col] = (bf16)0.0f;
    }
}

// ---------------------------------------------------------------------------
// Span head: num_spans = softmax(hx @ W_span^T + b_span).  One wave32 per row.
// ---------------------------------------------------------------------------
__global__ void span_head(const float* __restrict__ hx,
                          const float* __restrict__ Wspan,
                          const float* __restrict__ bspan,
                          float* __restrict__ out) {
    const int lane = threadIdx.x & 31;
    const int wave = threadIdx.x >> 5;
    const int row  = blockIdx.x * (blockDim.x >> 5) + wave;
    if (row >= BS) return;
    const float* hr = hx + (size_t)row * HID;

    float acc[NSPAN];
#pragma unroll
    for (int s = 0; s < NSPAN; ++s) acc[s] = 0.0f;
    for (int k = lane; k < HID; k += 32) {
        const float hv = hr[k];
#pragma unroll
        for (int s = 0; s < NSPAN; ++s) acc[s] += hv * Wspan[s * HID + k];
    }
#pragma unroll
    for (int s = 0; s < NSPAN; ++s)
        for (int off = 16; off > 0; off >>= 1) acc[s] += __shfl_down(acc[s], off, 32);

    if (lane == 0) {
        float m = -3.4e38f;
#pragma unroll
        for (int s = 0; s < NSPAN; ++s) { acc[s] += bspan[s]; m = fmaxf(m, acc[s]); }
        float sum = 0.0f;
#pragma unroll
        for (int s = 0; s < NSPAN; ++s) { acc[s] = __expf(acc[s] - m); sum += acc[s]; }
        const float inv = 1.0f / sum;
#pragma unroll
        for (int s = 0; s < NSPAN; ++s) out[(size_t)row * NSPAN + s] = acc[s] * inv;
    }
}

// ---------------------------------------------------------------------------
// Fused LSTM step (WMMA + async-LDS double buffering):
//   gates = Xcur @ Wcat^T (+bias), then cell nonlinearity.
// Block = 512 threads = 16 wave32s; block tile 64(M) x 64(N) x 4 gates.
// Per K-chunk (32): A tile 64x32 bf16 (4KB), B tile (4*64)x32 bf16 (16KB) are
// DMA'd into LDS with exactly 3 branch-free async-to-LDS issues per thread
// (1x b64 for A, 2x b128 for B).  Chunk k+1's DMA is issued BEFORE waiting
// on chunk k (s_wait_asynccnt 0x3 + in-order async completion).  All ten
// fragment ds_load_b128s are hoisted ahead of the 4-WMMA burst so dscnt
// waits stagger behind v_wmma_f32_16x16x32_bf16 execution.
// ---------------------------------------------------------------------------
__global__ __launch_bounds__(512)
void lstm_step(const bf16* __restrict__ Xcur,
               bf16* __restrict__ Xnxt,
               const bf16* __restrict__ Wcat,
               const float* __restrict__ biasg,
               float* __restrict__ c,
               float* __restrict__ h) {
    __shared__ bf16 smA[2][64][LDSTR];    // 2 x 5.0 KB
    __shared__ bf16 smB[2][256][LDSTR];   // 2 x 20.0 KB

    const int tid  = threadIdx.x;
    const int lane = tid & 31;
    const int wave = tid >> 5;
    const int mw = wave >> 2;            // 0..3
    const int nw = wave & 3;             // 0..3
    const int m0b = blockIdx.x * 64;     // batch-row base of this block
    const int n0b = blockIdx.y * 64;     // hidden-col base of this block
    const int l16  = lane & 15;
    const int half = lane >> 4;          // K sub-block select per ISA layout

    v8f acc0 = {}, acc1 = {}, acc2 = {}, acc3 = {};   // i, f, g, o

    // Branch-free, wave-uniform DMA issue: every thread issues exactly
    // 1x async b64 (A tile) + 2x async b128 (B tile) = 3 per chunk.
    auto issue_chunk = [&](int kb, int buf) {
        const unsigned aBase = (unsigned)(uintptr_t)&smA[buf][0][0];
        const unsigned bBase = (unsigned)(uintptr_t)&smB[buf][0][0];
        const int k0 = kb * 32;
        {   // A tile: 64 rows x 64B -> 512 b64 segments, one per thread
            const int row = tid >> 3, seg = tid & 7;
            const bf16* g = Xcur + (size_t)(m0b + row) * KX + k0 + seg * 4;
            async_b64(aBase + (unsigned)(row * (LDSTR * 2) + seg * 8), g);
        }
        {   // B tile rows 0..127 (gates 0-1): b128 segment tid
            const int row = tid >> 2, seg = tid & 3;
            const bf16* g = Wcat +
                (size_t)((row >> 6) * HID + n0b + (row & 63)) * KX + k0 + seg * 8;
            async_b128(bBase + (unsigned)(row * (LDSTR * 2) + seg * 16), g);
        }
        {   // B tile rows 128..255 (gates 2-3): b128 segment tid+512
            const int s = tid + 512;
            const int row = s >> 2, seg = s & 3;
            const bf16* g = Wcat +
                (size_t)((row >> 6) * HID + n0b + (row & 63)) * KX + k0 + seg * 8;
            async_b128(bBase + (unsigned)(row * (LDSTR * 2) + seg * 16), g);
        }
    };

    issue_chunk(0, 0);                   // prologue: prime buffer 0

    const int arow = mw * 16 + l16;
    const int brow0 = 0 * 64 + nw * 16 + l16;
    const int brow1 = 1 * 64 + nw * 16 + l16;
    const int brow2 = 2 * 64 + nw * 16 + l16;
    const int brow3 = 3 * 64 + nw * 16 + l16;

    for (int kb = 0; kb < NK; ++kb) {
        const int cur = kb & 1;
        // Issue next chunk's DMA first (its target buffer was last read two
        // barriers ago), then wait only until the current chunk's 3 in-order
        // DMAs have retired — next chunk keeps streaming during compute.
        if (kb + 1 < NK) {
            issue_chunk(kb + 1, cur ^ 1);
            wait_async_le3();
        } else {
            wait_async0();
        }
        __syncthreads();                 // all waves' DMAs landed -> tile ready

        // Hoist ALL fragment loads (10x ds_load_b128) ahead of the WMMA burst.
        FragBF a, b0, b1, b2, b3;
        a.q[0]  = *(const uint4*)&smA[cur][arow][half * 8];
        a.q[1]  = *(const uint4*)&smA[cur][arow][16 + half * 8];
        b0.q[0] = *(const uint4*)&smB[cur][brow0][half * 8];
        b0.q[1] = *(const uint4*)&smB[cur][brow0][16 + half * 8];
        b1.q[0] = *(const uint4*)&smB[cur][brow1][half * 8];
        b1.q[1] = *(const uint4*)&smB[cur][brow1][16 + half * 8];
        b2.q[0] = *(const uint4*)&smB[cur][brow2][half * 8];
        b2.q[1] = *(const uint4*)&smB[cur][brow2][16 + half * 8];
        b3.q[0] = *(const uint4*)&smB[cur][brow3][half * 8];
        b3.q[1] = *(const uint4*)&smB[cur][brow3][16 + half * 8];

        acc0 = __builtin_amdgcn_wmma_f32_16x16x32_bf16(
            false, a.v, false, b0.v, (short)0, acc0, false, false);
        acc1 = __builtin_amdgcn_wmma_f32_16x16x32_bf16(
            false, a.v, false, b1.v, (short)0, acc1, false, false);
        acc2 = __builtin_amdgcn_wmma_f32_16x16x32_bf16(
            false, a.v, false, b2.v, (short)0, acc2, false, false);
        acc3 = __builtin_amdgcn_wmma_f32_16x16x32_bf16(
            false, a.v, false, b3.v, (short)0, acc3, false, false);

        __syncthreads();                 // all waves done reading buf `cur`
    }

    // Epilogue: C/D layout -> element r is row (m0 + half*8 + r), col (n0 + l16)
    const int m0 = m0b + mw * 16;
    const int n  = n0b + nw * 16 + l16;
    const float bi  = biasg[0 * HID + n];
    const float bf_ = biasg[1 * HID + n];
    const float bg  = biasg[2 * HID + n];
    const float bo  = biasg[3 * HID + n];
#pragma unroll
    for (int r = 0; r < 8; ++r) {
        const int brow_ = m0 + half * 8 + r;
        const size_t idx = (size_t)brow_ * HID + n;
        const float iv = sigm(acc0[r] + bi);
        const float fv = sigm(acc1[r] + bf_);
        const float gv = tanhf(acc2[r] + bg);
        const float ov = sigm(acc3[r] + bo);
        const float cn = fv * c[idx] + iv * gv;
        const float hn = ov * tanhf(cn);
        c[idx] = cn;
        h[idx] = hn;
        Xnxt[(size_t)brow_ * KX + INP + n] = (bf16)hn;   // bf16 h for next GEMMs
    }
}

// ---------------------------------------------------------------------------
// fcin GEMM (WMMA, direct L2 path): inp = relu(h @ W_fcin^T + b_fcin),
// bf16 out into Xnxt[:, :INP].  Block = 256 threads = 8 wave32s.
// Reads h (bf16) from Xnxt[:, INP:], writes Xnxt[:, :INP] (disjoint).
// ---------------------------------------------------------------------------
__global__ __launch_bounds__(256)
void fcin_gemm(bf16* __restrict__ Xnxt,
               const bf16* __restrict__ Wfc,
               const float* __restrict__ bfc) {
    const int lane = threadIdx.x & 31;
    const int wave = threadIdx.x >> 5;
    const int mw = wave >> 2;            // 0..1
    const int nw = wave & 3;             // 0..3
    const int m0 = blockIdx.x * 32 + mw * 16;
    const int n0 = blockIdx.y * 64 + nw * 16;
    const int l16  = lane & 15;
    const int half = lane >> 4;

    v8f acc = {};
    const bf16* __restrict__ arow = Xnxt + (size_t)(m0 + l16) * KX + INP;  // h region
    const bf16* __restrict__ brow = Wfc + (size_t)(n0 + l16) * HID;

    for (int kb = 0; kb < HID; kb += 32) {
        FragBF a, b;
        a.q[0] = *(const uint4*)(arow + kb + half * 8);
        a.q[1] = *(const uint4*)(arow + kb + 16 + half * 8);
        b.q[0] = *(const uint4*)(brow + kb + half * 8);
        b.q[1] = *(const uint4*)(brow + kb + 16 + half * 8);
        acc = __builtin_amdgcn_wmma_f32_16x16x32_bf16(false, a.v, false, b.v,
                                                      (short)0, acc, false, false);
    }

    const int n = n0 + l16;
    const float bias = bfc[n];
#pragma unroll
    for (int r = 0; r < 8; ++r) {
        const int brow_ = m0 + half * 8 + r;
        float v = acc[r] + bias;
        v = v > 0.0f ? v : 0.0f;                      // ReLU
        Xnxt[(size_t)brow_ * KX + n] = (bf16)v;
    }
}

// ---------------------------------------------------------------------------
// fcout head: out[b,t,:] = softmax(h[b] @ W_fcout^T + b_fcout). One wave per row.
// ---------------------------------------------------------------------------
__global__ void fcout_softmax(const float* __restrict__ h,
                              const float* __restrict__ Wout,
                              const float* __restrict__ bout,
                              float* __restrict__ out, int t) {
    const int lane = threadIdx.x & 31;
    const int wave = threadIdx.x >> 5;
    const int row  = blockIdx.x * (blockDim.x >> 5) + wave;
    if (row >= BS) return;
    const float* hr = h + (size_t)row * HID;

    float a0 = 0.0f, a1 = 0.0f;
    for (int k = lane; k < HID; k += 32) {
        const float hv = hr[k];
        a0 += hv * Wout[k];
        a1 += hv * Wout[HID + k];
    }
    for (int off = 16; off > 0; off >>= 1) {
        a0 += __shfl_down(a0, off, 32);
        a1 += __shfl_down(a1, off, 32);
    }
    if (lane == 0) {
        a0 += bout[0];
        a1 += bout[1];
        const float m  = fmaxf(a0, a1);
        const float e0 = __expf(a0 - m);
        const float e1 = __expf(a1 - m);
        const float inv = 1.0f / (e0 + e1);
        const size_t o = (size_t)BS * NSPAN + ((size_t)row * TSTEPS + t) * NOUT;
        out[o]     = e0 * inv;
        out[o + 1] = e1 * inv;
    }
}

// ---------------------------------------------------------------------------
// Launch
// ---------------------------------------------------------------------------
extern "C" void kernel_launch(void* const* d_in, const int* in_sizes, int n_in,
                              void* d_out, int out_size, void* d_ws, size_t ws_size,
                              hipStream_t stream) {
    (void)in_sizes; (void)n_in; (void)out_size; (void)ws_size;

    const float* hx      = (const float*)d_in[0];
    const float* cx      = (const float*)d_in[1];
    const float* W_ih    = (const float*)d_in[2];
    const float* W_hh    = (const float*)d_in[3];
    const float* b_ih    = (const float*)d_in[4];
    const float* b_hh    = (const float*)d_in[5];
    const float* W_fcin  = (const float*)d_in[6];
    const float* b_fcin  = (const float*)d_in[7];
    const float* W_fcout = (const float*)d_in[8];
    const float* b_fcout = (const float*)d_in[9];
    const float* W_span  = (const float*)d_in[10];
    const float* b_span  = (const float*)d_in[11];
    float* out = (float*)d_out;

    // Workspace bump allocator (256B aligned)
    char* w = (char*)d_ws;
    auto bump = [&](size_t bytes) -> void* {
        void* p = (void*)w;
        w += (bytes + 255) & ~(size_t)255;
        return p;
    };
    bf16*  Wcat  = (bf16*) bump((size_t)G4 * KX * sizeof(bf16));    // 12 MB
    bf16*  Wfc   = (bf16*) bump((size_t)INP * HID * sizeof(bf16));  //  1 MB
    float* biasg = (float*)bump((size_t)G4 * sizeof(float));
    bf16*  X0    = (bf16*) bump((size_t)BS * KX * sizeof(bf16));    // 12 MB
    bf16*  X1    = (bf16*) bump((size_t)BS * KX * sizeof(bf16));    // 12 MB
    float* hbuf  = (float*)bump((size_t)BS * HID * sizeof(float));  // 16 MB
    float* cbuf  = (float*)bump((size_t)BS * HID * sizeof(float));  // 16 MB

    prep_weights<<<512, 256, 0, stream>>>(W_ih, W_hh, W_fcin, b_ih, b_hh,
                                          Wcat, Wfc, biasg);
    init_state<<<512, 256, 0, stream>>>(hx, cx, hbuf, cbuf, X0);
    span_head<<<BS / 8, 256, 0, stream>>>(hx, W_span, b_span, out);

    bf16* Xc = X0;
    bf16* Xn = X1;
    for (int t = 0; t < TSTEPS; ++t) {
        lstm_step<<<dim3(BS / 64, HID / 64), 512, 0, stream>>>(
            Xc, Xn, Wcat, biasg, cbuf, hbuf);
        fcin_gemm<<<dim3(BS / 32, INP / 64), 256, 0, stream>>>(Xn, Wfc, b_fcin);
        fcout_softmax<<<BS / 8, 256, 0, stream>>>(hbuf, W_fcout, b_fcout, out, t);
        bf16* tmp = Xc; Xc = Xn; Xn = tmp;
    }
}
